// MultiHeadAttention_54958401520161
// MI455X (gfx1250) — compile-verified, matrix operand order checked by reading, not run on hardware
//
#include <hip/hip_runtime.h>

#define DM    32      // d_model
#define NH    8       // heads
#define CPH   4       // depth per head
#define QS    24
#define FL    8
#define BATCH 2
#define CIN   64
#define HH    32
#define WW    192
#define TT    192
#define TP    208     // T + 2*FL
#define MB    8       // T / QS
#define FF    40      // 2*FL + QS
#define NCOL  (HH*FF) // 1280
#define NROW  (HH*QS) // 768

typedef __attribute__((ext_vector_type(2)))  float    v2f;
typedef __attribute__((ext_vector_type(8)))  float    v8f;
typedef __attribute__((ext_vector_type(16))) _Float16 v16h;
typedef __attribute__((ext_vector_type(4)))  unsigned int u32x4;
typedef __attribute__((ext_vector_type(8)))  int      i32x8;
typedef __attribute__((ext_vector_type(4)))  int      i32x4;

// ---------------------------------------------------------------------------
// TDM: async DMA of a 3-D tile (40 x 32 x 4 f32; strides 192 / 6144 elements)
// from global into contiguous LDS. Tensor dims are set huge so the address
// math stays purely linear (this is what faithfully reproduces the reference
// as_strided row-slip past column 192). Tracked by TENSORcnt.
// D# bitfields per CDNA5 ISA ch.8 (group0: count/lds/global/type, group1:
// data_size + tile dims + strides, group2: higher dims, group3: unused).
// ---------------------------------------------------------------------------
__device__ __forceinline__ void tdm_load_tile(const float* gsrc, float* ldst) {
  unsigned lds_addr = (unsigned)(unsigned long long)(uintptr_t)ldst;  // low 32 bits = LDS byte offset
  unsigned long long ga = (unsigned long long)(uintptr_t)gsrc;
  u32x4 g0;
  g0[0] = 1u;                                   // count=1 (valid), user mode, no gather
  g0[1] = lds_addr;                             // lds_addr[63:32 of D#]
  g0[2] = (unsigned)(ga & 0xFFFFFFFFu);         // global_addr lo
  g0[3] = (unsigned)(ga >> 32) | (2u << 30);    // global_addr hi | type=2 ("image")
  i32x8 g1;
  g1[0] = 0x00020000;                           // workgroup_mask=0, data_size=2 (4B)
  g1[1] = 0;                                    // no atomic barrier; tensor_dim0[15:0]=0
  g1[2] = 0x0010;                               // tensor_dim0 = 1M (hi half); tensor_dim1[15:0]=0
  g1[3] = 0x0010 | (FF << 16);                  // tensor_dim1 = 1M (hi half); tile_dim0 = 40
  g1[4] = HH | (CPH << 16);                     // tile_dim1 = 32; tile_dim2 = 4
  g1[5] = TT;                                   // tensor_dim0_stride = 192
  g1[6] = (HH * TT) << 16;                      // tensor_dim1_stride[15:0] = 6144
  g1[7] = 0;                                    // tensor_dim1_stride hi
  i32x4 g2;
  g2[0] = 1 << 20;                              // tensor_dim2 (large)
  g2[1] = 1;                                    // tensor_dim3
  g2[2] = 0;                                    // tensor_dim2_stride lo
  g2[3] = 0;                                    // stride hi; tile_dim3 = 0 (3-D tile)
  i32x4 g3 = (i32x4){0, 0, 0, 0};
#if defined(__clang_major__) && __clang_major__ >= 23
  i32x8 g4 = (i32x8){0, 0, 0, 0, 0, 0, 0, 0};
  __builtin_amdgcn_tensor_load_to_lds(g0, g1, g2, g3, g4, 0);
#else
  __builtin_amdgcn_tensor_load_to_lds(g0, g1, g2, g3, 0);
#endif
}

// ---------------------------------------------------------------------------
// Kernel 1: fused Q/K/V 3x3 SAME conv as WMMA implicit GEMM.
//   M = B*H*W pixels (16-pixel tiles, never cross an image row: 16 | 192)
//   N = 96 output channels (q|k|v), 6 tiles of 16;  K = Cin*9 = 576 (ci,ky,kx)
// One wave per M-tile; 144 K-steps of v_wmma_f32_16x16x4_f32 per N-tile.
// ---------------------------------------------------------------------------
__global__ void __launch_bounds__(128) qkv_conv_wmma(
    const float* __restrict__ x,
    const float* __restrict__ qw, const float* __restrict__ qb,
    const float* __restrict__ kw, const float* __restrict__ kb,
    const float* __restrict__ vw, const float* __restrict__ vb,
    float* __restrict__ qbuf, float* __restrict__ kpad, float* __restrict__ vpad) {
  const int wave = threadIdx.x >> 5;
  const int lane = threadIdx.x & 31;
  const int ln   = lane & 15;
  const bool lo  = lane < 16;
  const int koff = lo ? 0 : 2;   // A/B K offset for hi half
  const int roff = lo ? 0 : 8;   // D row offset for hi half

  const int mt = blockIdx.x * 4 + wave;  // M-tile 0..767
  const int p0 = mt * 16;
  const int t0 = p0 % WW;
  const int y  = (p0 / WW) % HH;
  const int b  = p0 / (WW * HH);
  const int t  = t0 + ln;

  __builtin_prefetch(x + ((size_t)(b * CIN) * HH + y) * WW + t0, 0, 1);

  const float* const wp[3] = {qw, kw, vw};

  v8f acc[6];
#pragma unroll
  for (int n = 0; n < 6; ++n) acc[n] = (v8f){};

  for (int kb4 = 0; kb4 < CIN * 9; kb4 += 4) {
    v2f A;
#pragma unroll
    for (int u = 0; u < 2; ++u) {
      int k  = kb4 + koff + u;
      int ci = k / 9;
      int r9 = k - ci * 9;
      int ky = r9 / 3;
      int kx = r9 - ky * 3;
      int yy = y + ky - 1;
      int tt = t + kx - 1;
      float xv = 0.f;
      if (yy >= 0 && yy < HH && tt >= 0 && tt < WW)
        xv = x[((b * CIN + ci) * HH + yy) * WW + tt];
      A[u] = xv;
    }
#pragma unroll
    for (int n = 0; n < 6; ++n) {
      const float* w = wp[n >> 1];
      int ch = (n & 1) * 16 + ln;
      const float* wk = w + ch * (CIN * 9) + kb4 + koff;  // contiguous K pair
      v2f Bv;
      Bv.x = wk[0];
      Bv.y = wk[1];
      acc[n] = __builtin_amdgcn_wmma_f32_16x16x4_f32(false, A, false, Bv,
                                                     (short)0, acc[n], false, false);
    }
  }

#pragma unroll
  for (int n = 0; n < 6; ++n) {
    const int which = n >> 1;              // 0=q, 1=k, 2=v
    const int ch = (n & 1) * 16 + ln;
#pragma unroll
    for (int r = 0; r < 8; ++r) {
      int tt = t0 + r + roff;
      float val = acc[n][r];
      if (which == 0) {
        qbuf[((b * DM + ch) * HH + y) * WW + tt] = (val + qb[ch]) * 0.5f;
      } else if (which == 1) {
        kpad[((b * DM + ch) * HH + y) * TP + FL + tt] = val + kb[ch];
      } else {
        vpad[((b * DM + ch) * HH + y) * TP + FL + tt] = val + vb[ch];
      }
    }
  }

  // Zero the time padding once per (b,y) image row: the t0==0 wave does all 32 ch.
  if (t0 == 0) {
    int ch = lane;  // 0..31
    int base = ((b * DM + ch) * HH + y) * TP;
#pragma unroll
    for (int f = 0; f < FL; ++f) {
      kpad[base + f] = 0.f;            vpad[base + f] = 0.f;
      kpad[base + FL + TT + f] = 0.f;  vpad[base + FL + TT + f] = 0.f;
    }
  }
}

// ---------------------------------------------------------------------------
// Kernel 2: local attention per group g=(b,h,m). k2/v2 staged by TDM async
// DMA (overlapped with the q-tile loads). QK^T via V_WMMA_F32_16X16X4_F32
// computing S^T tiles, softmax with one cross-half shfl per row, P@V via
// V_WMMA_F32_16X16X32_F16 reading P straight from LDS in f16 A-layout order.
// ---------------------------------------------------------------------------
__global__ void __launch_bounds__(64) attn_kernel(const float* __restrict__ qbuf,
                                                  const float* __restrict__ kpad,
                                                  const float* __restrict__ vpad,
                                                  float* __restrict__ obuf) {
  extern __shared__ float smem[];
  float* k2s      = smem;                  // [4][NCOL]
  float* v2s      = k2s + 4 * NCOL;        // [4][NCOL]
  float* colbias  = v2s + 4 * NCOL;        // [NCOL]
  float* stripAll = colbias + NCOL;        // [2][16][NCOL]

  const int g = blockIdx.y;
  const int m = g % MB;
  const int h = (g / MB) % NH;
  const int b = g / (MB * NH);

  const int tid  = threadIdx.x;
  const int wave = tid >> 5;
  const int lane = tid & 31;
  const int ln   = lane & 15;
  const bool lo  = lane < 16;
  const int roff = lo ? 0 : 8;
  float* strip = stripAll + wave * 16 * NCOL;

  // Issue the k2/v2 DMA first (wave 0), then overlap with q-tile loads.
  // Tile base replays the reference as_strided linear index:
  // L = ((b*nh+h)*c+kk)*H*T + y*T + m*qs + f  (kk,y,f covered by the 3-D tile).
  const int gbase = ((b * NH + h) * CPH) * HH * TT + m * QS;
  if (wave == 0) {
    tdm_load_tile(kpad + gbase, k2s);
    tdm_load_tile(vpad + gbase, v2s);
  }

  // qq^T as B-matrix (4x16): N = S-row on lane, K on VGPR+half.
  const int rt   = blockIdx.x * 2 + wave;  // row tile 0..47
  const int row0 = rt * 16;
  int rowq = row0 + ln;
  int yq   = rowq / QS;
  int iq   = rowq - yq * QS;
  int kkq  = lo ? 0 : 2;
  int qidx = ((b * DM + h * CPH + kkq) * HH + yq) * WW + m * QS + iq;
  v2f Bq;
  Bq.x = qbuf[qidx];
  Bq.y = qbuf[qidx + HH * WW];  // next depth channel

  if (wave == 0) __builtin_amdgcn_s_wait_tensorcnt(0);
  __syncthreads();

  // Zero-key-column bias table (faithful -1e-9).
  for (int col = tid; col < NCOL; col += blockDim.x) {
    float s = __builtin_fabsf(k2s[col]) + __builtin_fabsf(k2s[NCOL + col]) +
              __builtin_fabsf(k2s[2 * NCOL + col]) + __builtin_fabsf(k2s[3 * NCOL + col]);
    colbias[col] = (s == 0.f) ? -1e-9f : 0.f;
  }
  __syncthreads();

  // Phase 1: S^T tiles via f32 WMMA (K=4 exact). This lane's 8 results are the
  // CONSECUTIVE columns ct*16+roff .. +roff+7 of row ln -> vector LDS store.
  float rm = -3.0e38f;
  for (int ct = 0; ct < NCOL / 16; ++ct) {
    int col = ct * 16 + ln;  // k2^T A-matrix: M = S-col on lane, K on VGPR+half
    v2f A;
    if (lo) { A.x = k2s[col];            A.y = k2s[NCOL + col];     }
    else    { A.x = k2s[2 * NCOL + col]; A.y = k2s[3 * NCOL + col]; }
    v8f Cz = {};
    v8f S = __builtin_amdgcn_wmma_f32_16x16x4_f32(false, A, false, Bq,
                                                  (short)0, Cz, false, false);
    v8f cb = *(const v8f*)&colbias[ct * 16 + roff];
    v8f Sv;
#pragma unroll
    for (int r = 0; r < 8; ++r) {
      float s = S[r] + cb[r];
      Sv[r] = s;
      rm = fmaxf(rm, s);
    }
    *(v8f*)&strip[ln * NCOL + ct * 16 + roff] = Sv;
  }
  rm = fmaxf(rm, __shfl_xor(rm, 16));
  __syncthreads();

  // Phase 2: exp + row sum, float4 LDS traffic (half the columns per lane half).
  float ls = 0.f;
  const int cbase = lo ? 0 : (NCOL / 2);
  for (int j = 0; j < NCOL / 2; j += 4) {
    float4 sv = *(const float4*)&strip[ln * NCOL + cbase + j];
    float4 pv;
    pv.x = __expf(sv.x - rm);
    pv.y = __expf(sv.y - rm);
    pv.z = __expf(sv.z - rm);
    pv.w = __expf(sv.w - rm);
    *(float4*)&strip[ln * NCOL + cbase + j] = pv;
    ls += (pv.x + pv.y) + (pv.z + pv.w);
  }
  ls += __shfl_xor(ls, 16);
  float inv = 1.0f / ls;
  __syncthreads();

  // Phase 3: O = P @ v2 via f16 WMMA; strip is already in f16 A-layout order.
  v8f O = {};
  for (int kc = 0; kc < NCOL / 32; ++kc) {
    int kb2 = kc * 32;
    v16h Ah, Bh;
#pragma unroll
    for (int j = 0; j < 8; ++j) {
      int base = kb2 + ((j < 4) ? 0 : 16) + roff + 2 * (j & 3);
      Ah[2 * j]     = (_Float16)strip[ln * NCOL + base];
      Ah[2 * j + 1] = (_Float16)strip[ln * NCOL + base + 1];
      float b0 = 0.f, b1 = 0.f;
      if (ln < CPH) {  // only 4 valid output columns
        b0 = v2s[ln * NCOL + base];
        b1 = v2s[ln * NCOL + base + 1];
      }
      Bh[2 * j]     = (_Float16)b0;
      Bh[2 * j + 1] = (_Float16)b1;
    }
    O = __builtin_amdgcn_wmma_f32_16x16x32_f16(false, Ah, false, Bh,
                                               (short)0, O, false, false);
  }

  // Phase 4: normalize per row (broadcast 1/l across VGPR rows) and store.
#pragma unroll
  for (int r = 0; r < 8; ++r) {
    float invr = __shfl(inv, r + roff, 16);
    float ov = O[r] * invr;
    if (ln < CPH) {
      int orow = row0 + r + roff;
      int y2 = orow / QS;
      int i2 = orow - y2 * QS;
      obuf[((b * DM + h * CPH + ln) * HH + y2) * WW + m * QS + i2] = ov;
    }
  }
}

// ---------------------------------------------------------------------------
// Kernel 3: output 3x3 SAME conv (32 -> 64), WMMA implicit GEMM.
//   M = B*H*W pixels, N = 64 (4 tiles), K = 32*9 = 288.
// ---------------------------------------------------------------------------
__global__ void __launch_bounds__(128) out_conv_wmma(const float* __restrict__ obuf,
                                                     const float* __restrict__ ow,
                                                     float* __restrict__ out) {
  const int wave = threadIdx.x >> 5;
  const int lane = threadIdx.x & 31;
  const int ln   = lane & 15;
  const bool lo  = lane < 16;
  const int koff = lo ? 0 : 2;
  const int roff = lo ? 0 : 8;

  const int mt = blockIdx.x * 4 + wave;
  const int p0 = mt * 16;
  const int t0 = p0 % WW;
  const int y  = (p0 / WW) % HH;
  const int b  = p0 / (WW * HH);
  const int t  = t0 + ln;

  __builtin_prefetch(obuf + ((size_t)(b * DM) * HH + y) * WW + t0, 0, 1);

  v8f acc[4];
#pragma unroll
  for (int n = 0; n < 4; ++n) acc[n] = (v8f){};

  for (int kb4 = 0; kb4 < DM * 9; kb4 += 4) {
    v2f A;
#pragma unroll
    for (int u = 0; u < 2; ++u) {
      int k  = kb4 + koff + u;
      int ci = k / 9;
      int r9 = k - ci * 9;
      int ky = r9 / 3;
      int kx = r9 - ky * 3;
      int yy = y + ky - 1;
      int tt = t + kx - 1;
      float xv = 0.f;
      if (yy >= 0 && yy < HH && tt >= 0 && tt < WW)
        xv = obuf[((b * DM + ci) * HH + yy) * WW + tt];
      A[u] = xv;
    }
#pragma unroll
    for (int n = 0; n < 4; ++n) {
      int ch = n * 16 + ln;
      const float* wk = ow + ch * (DM * 9) + kb4 + koff;
      v2f Bv;
      Bv.x = wk[0];
      Bv.y = wk[1];
      acc[n] = __builtin_amdgcn_wmma_f32_16x16x4_f32(false, A, false, Bv,
                                                     (short)0, acc[n], false, false);
    }
  }

#pragma unroll
  for (int n = 0; n < 4; ++n) {
    int ch = n * 16 + ln;
#pragma unroll
    for (int r = 0; r < 8; ++r) {
      int tt = t0 + r + roff;
      out[((b * CIN + ch) * HH + y) * WW + tt] = acc[n][r];
    }
  }
}

extern "C" void kernel_launch(void* const* d_in, const int* in_sizes, int n_in,
                              void* d_out, int out_size, void* d_ws, size_t ws_size,
                              hipStream_t stream) {
  (void)in_sizes; (void)n_in; (void)out_size; (void)ws_size;
  const float* x  = (const float*)d_in[0];
  const float* qw = (const float*)d_in[1];
  const float* qb = (const float*)d_in[2];
  const float* kw = (const float*)d_in[3];
  const float* kb = (const float*)d_in[4];
  const float* vw = (const float*)d_in[5];
  const float* vb = (const float*)d_in[6];
  const float* ow = (const float*)d_in[7];

  float* ws   = (float*)d_ws;
  float* qbuf = ws;                              // B*32*H*W   = 393216
  float* kpad = qbuf + BATCH * DM * HH * WW;     // B*32*H*TP  = 425984
  float* vpad = kpad + BATCH * DM * HH * TP;
  float* obuf = vpad + BATCH * DM * HH * TP;     // B*32*H*W

  const int mtiles = BATCH * HH * WW / 16;       // 768 M-tiles, 4 waves/block
  qkv_conv_wmma<<<mtiles / 4, 128, 0, stream>>>(
      x, qw, qb, kw, kb, vw, vb, qbuf, kpad, vpad);

  // LDS: k2(20K) + v2(20K) + colbias(5K) + 2 score strips(160K) = 205KB of 320KB
  size_t shmem = (size_t)(4 * NCOL + 4 * NCOL + NCOL + 2 * 16 * NCOL) * sizeof(float);
  attn_kernel<<<dim3(24, BATCH * NH * MB), 64, shmem, stream>>>(qbuf, kpad, vpad, obuf);

  out_conv_wmma<<<mtiles / 4, 128, 0, stream>>>(obuf, ow, (float*)d_out);
}
